// MPSLayer_55533927137910
// MI455X (gfx1250) — matrix operand order repaired and checked by reading, not compiled, not run on hardware
//
#include <hip/hip_runtime.h>
#include <hip/hip_bf16.h>
#include <stdint.h>

// ---- problem constants (match reference) ----
#define B_TOT 512
#define NPIX  784
#define NC    391          // cores per side (HALF-1)
#define D     64
#define PDIM  2
#define LBL   10

// ---- tiling ----
#define BT      32         // batch rows per workgroup
#define LC_ST   68         // f32 row stride of l-buffer (bank-conflict free, 16B aligned)
#define WT_ST32 36         // u32 row stride of staged-core buffer (144B: 128B data + 16B pad)
#define CORE_ELEMS   (D * PDIM * D)                  // 8192 f32 per core
#define STAGED_BYTES (128 * 64 * 2)                  // 16384 B bf16 per staged core

typedef __attribute__((ext_vector_type(16))) __bf16 v16bf;
typedef __attribute__((ext_vector_type(8)))  float  v8f;
typedef __attribute__((ext_vector_type(4)))  unsigned int v4u;
typedef __attribute__((ext_vector_type(8)))  int    v8i;
typedef __attribute__((ext_vector_type(4)))  int    v4i;

union AFrag { v16bf v; __bf16 h[16]; };
union BFrag { v16bf v; uint4  q[2];  };
union PK2   { __bf16 h[2]; uint32_t u; };

#if __has_include(<hip/amd_detail/amd_gfx1250_TDM.h>)
#define MPS_TDM_6ARG 1
#else
#define MPS_TDM_6ARG 0
#endif

// ----------------------------------------------------------------------------
// TDM issue: DMA one 16-row (2 KB) slice of a staged bf16 core into LDS with
// 16B padding after every 128B row (matches WT_ST32 layout).
// D# per ISA cdna5 §8.3/8.4: group0 = {count|flags, lds_addr, gaddr_lo,
// gaddr_hi|type=2}; group1 = data_size=3(8B), pad_enable, pad_interval=32DW,
// pad_amount=4DW, tensor_dim0=tile_dim0=stride=256 (8B units), dim1=1.
// ----------------------------------------------------------------------------
__device__ __forceinline__ void tdm_issue_slice(const uint8_t* gsrc, void* ldsdst)
{
    const uint64_t ga = (uint64_t)(uintptr_t)gsrc;
    v4u g0;
    g0[0] = 1u;                                          // count=1, user mode
    g0[1] = (uint32_t)(uintptr_t)ldsdst;                 // LDS byte address
    g0[2] = (uint32_t)ga;                                // global addr [31:0]
    g0[3] = (uint32_t)((ga >> 32) & 0x01FFFFFFu) | (2u << 30);  // [56:32] | type=2
    v8i g1;
    g1[0] = (int)((3u << 16) | (1u << 20) | (4u << 22) | (3u << 25));
    //            data_size=8B  pad_en      pad_int=32DW  pad_amt=4DW
    g1[1] = (int)(256u << 16);   // tensor_dim0 = 256 (bits 79:48)
    g1[2] = (int)(1u << 16);     // tensor_dim1 = 1   (bits 111:80)
    g1[3] = (int)(256u << 16);   // tile_dim0   = 256 (bits 127:112)
    g1[4] = 1;                   // tile_dim1   = 1   (bits 143:128)
    g1[5] = 256;                 // tensor_dim0_stride = 256 (bits 207:160)
    g1[6] = 0;
    g1[7] = 0;
    v4i g2 = {0, 0, 0, 0};
    v4i g3 = {0, 0, 0, 0};
#if MPS_TDM_6ARG
    v8i g4 = {0, 0, 0, 0, 0, 0, 0, 0};
    __builtin_amdgcn_tensor_load_to_lds(g0, g1, g2, g3, g4, 0);
#else
    __builtin_amdgcn_tensor_load_to_lds(g0, g1, g2, g3, 0);
#endif
}

// ============================================================================
// Prep kernel: convert all cores f32->bf16 into workspace, pre-swizzled into
// the exact "B-transposed" layout the chain kernel stages (left: row=(p,e),
// K=d; right: row=(p,d), K=e). Embarrassingly parallel, bandwidth bound.
// grid = 2*NC blocks of 256 threads (one block per (core, side)).
// ============================================================================
__global__ __launch_bounds__(256) void mps_prep_kernel(
    const float* __restrict__ coresL, const float* __restrict__ coresR,
    uint8_t* __restrict__ wsL, uint8_t* __restrict__ wsR)
{
    const int s    = blockIdx.x >> 1;
    const int side = blockIdx.x & 1;
    const int t    = threadIdx.x;
    const float* g = (side ? coresR : coresL) + (size_t)s * CORE_ELEMS;
    uint32_t* dst  = (uint32_t*)((side ? wsR : wsL) + (size_t)s * STAGED_BYTES);

    const int c = t & 127;       // destination row
    const int h = t >> 7;        // K half (32 values)
    float v[32];
    if (side == 0) {
        // row c=(p,e)=p*64+e, K=d: column gather, coalesced across lanes
        const float* src = g + c;
#pragma unroll
        for (int j = 0; j < 32; ++j) v[j] = src[(size_t)(h * 32 + j) * 128];
    } else {
        // row c=(p,d)=p*64+d, K=e: contiguous reads
        const float* src = g + (size_t)(c & 63) * 128 + (c >> 6) * 64 + h * 32;
#pragma unroll
        for (int j = 0; j < 32; ++j) v[j] = src[j];
    }
    uint32_t* d = dst + c * 32 + h * 16;
#pragma unroll
    for (int j = 0; j < 16; ++j) {
        PK2 pk; pk.h[0] = (__bf16)v[2 * j]; pk.h[1] = (__bf16)v[2 * j + 1];
        d[j] = pk.u;
    }
}

// ============================================================================
// Kernel 1: full left (side=0) or right (side=1) sequential chain for a 32-row
// batch tile. grid.x = 2*(B_TOT/BT); block = 256 (8 waves).
// Per step: T[32,128] = L[32,64] @ W[64,128] via bf16 WMMA (f32 accum), then
// l'[b,e] = x0[b]*T[b,e] + x1[b]*T[b,64+e] combined in-register.
// USE_TDM=1: cores pre-staged bf16 in ws, moved global->LDS by TDM
//            (triple buffered, TENSORcnt-guarded).
// USE_TDM=0: f32 cores staged through registers (double buffered).
// ============================================================================
template <bool USE_TDM>
__global__ __launch_bounds__(256) void mps_chain_kernel(
    const float* __restrict__ x,
    const float* __restrict__ ledge,
    const float* __restrict__ coresL,
    const float* __restrict__ coresR,
    const float* __restrict__ redge,
    const uint8_t* __restrict__ wsL,   // staged bf16 cores (TDM path)
    const uint8_t* __restrict__ wsR,
    float* __restrict__ lout,
    float* __restrict__ rout)
{
    constexpr int NBUF = USE_TDM ? 3 : 2;
    __shared__ __align__(16) float    sLc[2][BT][LC_ST];
    __shared__ __align__(16) uint32_t sWT[NBUF][128][WT_ST32];
    __shared__ float sXB[2][PDIM][BT];

    const int t    = threadIdx.x;
    const int lane = t & 31;
    const int w    = t >> 5;
    const int side = blockIdx.x & 1;
    const int tile = blockIdx.x >> 1;
    const int b0   = tile * BT;

    const float*   gC   = side ? coresR : coresL;
    const uint8_t* gS   = side ? wsR    : wsL;
    const float*   edge = side ? redge  : ledge;
    float*         gOut = side ? rout   : lout;

    const int rt = w >> 2;          // 0..1 row tile
    const int ct = w & 3;           // 0..3 col tile (pairs with ct+4)
    const int n  = lane & 15;
    const int kh = lane >> 4;

    // ---------------- stage cores for the pipeline head ----------------
    if (USE_TDM) {
        // each wave DMAs its own 16-row slice; issue steps 0 and 1
        const int ci0 = side ? (NC - 1) : 0;
        const int ci1 = side ? (NC - 2) : 1;
        tdm_issue_slice(gS + (size_t)ci0 * STAGED_BYTES + w * 2048, &sWT[0][w * 16][0]);
        tdm_issue_slice(gS + (size_t)ci1 * STAGED_BYTES + w * 2048, &sWT[1][w * 16][0]);
        __builtin_amdgcn_s_wait_tensorcnt(1);   // slice for step 0 landed
    } else {
        const int ci = side ? (NC - 1) : 0;
        const float* gW = gC + (size_t)ci * CORE_ELEMS;
        const int c = t & 127;
        const int h = t >> 7;
        float val[32];
        if (side == 0) {
            const float* src = gW + c;
#pragma unroll
            for (int j = 0; j < 32; ++j) val[j] = src[(size_t)(h * 32 + j) * 128];
        } else {
            const float* src = gW + (size_t)(c & 63) * 128 + (c >> 6) * 64 + h * 32;
#pragma unroll
            for (int j = 0; j < 32; ++j) val[j] = src[j];
        }
        uint32_t* dst = &sWT[0][c][h * 16];
#pragma unroll
        for (int j = 0; j < 16; ++j) {
            PK2 pk; pk.h[0] = (__bf16)val[2 * j]; pk.h[1] = (__bf16)val[2 * j + 1];
            dst[j] = pk.u;
        }
    }

    // ---------------- stage x for step 0 ----------------
    if (t < PDIM * BT) {
        const int r = t & (BT - 1);
        const int p = t >> 5;
        const int pos = side ? (NPIX - 2) : 1;
        sXB[0][p][r] = x[(size_t)(b0 + r) * (NPIX * PDIM) + pos * PDIM + p];
    }

    // ---------------- init l0 / r0 from edge tensors ----------------
    {
        const int pos0 = side ? (NPIX - 1) : 0;
        const int r  = t >> 3;
        const int d8 = (t & 7) * 8;
        const float x0 = x[(size_t)(b0 + r) * (NPIX * PDIM) + pos0 * PDIM + 0];
        const float x1 = x[(size_t)(b0 + r) * (NPIX * PDIM) + pos0 * PDIM + 1];
#pragma unroll
        for (int j = 0; j < 8; ++j) {
            const int d = d8 + j;
            sLc[0][r][d] = x0 * edge[d * PDIM + 0] + x1 * edge[d * PDIM + 1];
        }
    }
    __syncthreads();

    // ---------------- main sequential chain ----------------
#pragma unroll 1
    for (int i = 0; i < NC; ++i) {
        const int cur  = i & 1;
        const int nxt  = cur ^ 1;
        const int rbuf = USE_TDM ? (i % 3) : cur;

        float val[32];
        const int c = t & 127;
        const int h = t >> 7;
        if (USE_TDM) {
            if (i + 2 < NC) {   // DMA core i+2 into the third buffer
                const int ci = side ? (NC - 3 - i) : (i + 2);
                tdm_issue_slice(gS + (size_t)ci * STAGED_BYTES + w * 2048,
                                &sWT[(i + 2) % 3][w * 16][0]);
            }
        } else {
            if (i + 1 < NC) {
                const int ci = side ? (NC - 2 - i) : (i + 1);
                const float* gW = gC + (size_t)ci * CORE_ELEMS;
                if (side == 0) {
                    const float* src = gW + c;
#pragma unroll
                    for (int j = 0; j < 32; ++j) val[j] = src[(size_t)(h * 32 + j) * 128];
                } else {
                    const float* src = gW + (size_t)(c & 63) * 128 + (c >> 6) * 64 + h * 32;
#pragma unroll
                    for (int j = 0; j < 32; ++j) val[j] = src[j];
                }
            }
            if (i + 2 < NC) {
                const int ci2 = side ? (NC - 3 - i) : (i + 2);
                __builtin_prefetch(gC + (size_t)ci2 * CORE_ELEMS + t * 32, 0, 0);
            }
        }

        // --- WMMA GEMM step i ---
        AFrag a0, a1;
        {
            const float* rp = &sLc[cur][rt * 16 + n][0];
            const float* p0 = rp + kh * 16;
#pragma unroll
            for (int j = 0; j < 8; ++j) a0.h[j]     = (__bf16)p0[j];
#pragma unroll
            for (int j = 0; j < 8; ++j) a0.h[8 + j] = (__bf16)p0[8 + j];
            const float* p1 = rp + 32 + kh * 16;
#pragma unroll
            for (int j = 0; j < 8; ++j) a1.h[j]     = (__bf16)p1[j];
#pragma unroll
            for (int j = 0; j < 8; ++j) a1.h[8 + j] = (__bf16)p1[8 + j];
        }
        BFrag b00, b01, b10, b11;
        {
            const uint32_t* r0 = &sWT[rbuf][ct * 16 + n][kh * 8];
            b00.q[0] = *(const uint4*)(r0 + 0);   b00.q[1] = *(const uint4*)(r0 + 4);
            b01.q[0] = *(const uint4*)(r0 + 16);  b01.q[1] = *(const uint4*)(r0 + 20);
            const uint32_t* r1 = &sWT[rbuf][(ct + 4) * 16 + n][kh * 8];
            b10.q[0] = *(const uint4*)(r1 + 0);   b10.q[1] = *(const uint4*)(r1 + 4);
            b11.q[0] = *(const uint4*)(r1 + 16);  b11.q[1] = *(const uint4*)(r1 + 20);
        }

        v8f c0 = {}, c1 = {};
        c0 = __builtin_amdgcn_wmma_f32_16x16x32_bf16(false, a0.v, false, b00.v, (short)0, c0, false, false);
        c0 = __builtin_amdgcn_wmma_f32_16x16x32_bf16(false, a1.v, false, b01.v, (short)0, c0, false, false);
        c1 = __builtin_amdgcn_wmma_f32_16x16x32_bf16(false, a0.v, false, b10.v, (short)0, c1, false, false);
        c1 = __builtin_amdgcn_wmma_f32_16x16x32_bf16(false, a1.v, false, b11.v, (short)0, c1, false, false);

        // in-register x-combine: l'[b, e=ct*16+n] = x0*T[b,(0,e)] + x1*T[b,(1,e)]
#pragma unroll
        for (int v = 0; v < 8; ++v) {
            const int rl = rt * 16 + v + 8 * kh;
            const float xv0 = sXB[cur][0][rl];
            const float xv1 = sXB[cur][1][rl];
            sLc[nxt][rl][ct * 16 + n] = xv0 * c0[v] + xv1 * c1[v];
        }

        // --- stage next x; finish next-core staging ---
        if (i + 1 < NC) {
            if (t < PDIM * BT) {
                const int r = t & (BT - 1);
                const int p = t >> 5;
                const int pos = side ? (NPIX - 3 - i) : (i + 2);
                sXB[nxt][p][r] = x[(size_t)(b0 + r) * (NPIX * PDIM) + pos * PDIM + p];
            }
            if (!USE_TDM) {
                uint32_t* dst = &sWT[nxt][c][h * 16];
#pragma unroll
                for (int j = 0; j < 16; ++j) {
                    PK2 pk; pk.h[0] = (__bf16)val[2 * j]; pk.h[1] = (__bf16)val[2 * j + 1];
                    dst[j] = pk.u;
                }
            }
        }
        if (USE_TDM) {
            // guarantee this wave's slice of W[i+1] has landed before the
            // barrier publishes all slices to all waves
            if (i + 2 < NC) __builtin_amdgcn_s_wait_tensorcnt(1);
            else            __builtin_amdgcn_s_wait_tensorcnt(0);
        }
        __syncthreads();
    }

    // ---------------- write final chain vector ----------------
    {
        const int fin = NC & 1;
        const int r  = t >> 3;
        const int d8 = (t & 7) * 8;
#pragma unroll
        for (int j = 0; j < 8; ++j)
            gOut[(size_t)(b0 + r) * D + d8 + j] = sLc[fin][r][d8 + j];
    }
}

// ============================================================================
// Kernel 2: out[b,l] = sum_d l[b,d] * sum_e lab[d,l,e] * r[b,e]
// ============================================================================
__global__ __launch_bounds__(32) void mps_label_kernel(
    const float* __restrict__ lab,
    const float* __restrict__ lvec,
    const float* __restrict__ rvec,
    float* __restrict__ out)
{
    const int b    = blockIdx.x;
    const int lane = threadIdx.x;
    __shared__ float sl[D], sr[D];
    sl[lane]      = lvec[(size_t)b * D + lane];
    sl[lane + 32] = lvec[(size_t)b * D + lane + 32];
    sr[lane]      = rvec[(size_t)b * D + lane];
    sr[lane + 32] = rvec[(size_t)b * D + lane + 32];
    __syncthreads();

    float acc[LBL];
#pragma unroll
    for (int l10 = 0; l10 < LBL; ++l10) acc[l10] = 0.0f;

    for (int dd = 0; dd < 2; ++dd) {
        const int d = lane + dd * 32;
        const float ld = sl[d];
        const float* labd = lab + (size_t)d * LBL * D;
        for (int l10 = 0; l10 < LBL; ++l10) {
            float s = 0.0f;
#pragma unroll
            for (int e = 0; e < D; ++e) s += labd[l10 * D + e] * sr[e];
            acc[l10] += ld * s;
        }
    }
#pragma unroll
    for (int off = 16; off >= 1; off >>= 1)
#pragma unroll
        for (int l10 = 0; l10 < LBL; ++l10)
            acc[l10] += __shfl_xor(acc[l10], off, 32);

    if (lane == 0)
        for (int l10 = 0; l10 < LBL; ++l10) out[(size_t)b * LBL + l10] = acc[l10];
}

// ============================================================================
extern "C" void kernel_launch(void* const* d_in, const int* in_sizes, int n_in,
                              void* d_out, int out_size, void* d_ws, size_t ws_size,
                              hipStream_t stream)
{
    (void)in_sizes; (void)n_in; (void)out_size;
    const float* x      = (const float*)d_in[0];
    const float* ledge  = (const float*)d_in[1];
    const float* coresL = (const float*)d_in[2];
    const float* lab    = (const float*)d_in[3];
    const float* coresR = (const float*)d_in[4];
    const float* redge  = (const float*)d_in[5];

    float* lout = (float*)d_ws;                      // 512*64 f32
    float* rout = lout + (size_t)B_TOT * D;          // 512*64 f32
    const size_t vecBytes  = (size_t)2 * B_TOT * D * sizeof(float);
    const size_t coreBytes = (size_t)NC * STAGED_BYTES;   // per side, bf16 staged
    uint8_t* wsL = (uint8_t*)d_ws + vecBytes;
    uint8_t* wsR = wsL + coreBytes;

    const bool use_tdm = (ws_size >= vecBytes + 2 * coreBytes);

    if (use_tdm) {
        mps_prep_kernel<<<dim3(2 * NC), dim3(256), 0, stream>>>(coresL, coresR, wsL, wsR);
        mps_chain_kernel<true><<<dim3(2 * (B_TOT / BT)), dim3(256), 0, stream>>>(
            x, ledge, coresL, coresR, redge, wsL, wsR, lout, rout);
    } else {
        mps_chain_kernel<false><<<dim3(2 * (B_TOT / BT)), dim3(256), 0, stream>>>(
            x, ledge, coresL, coresR, redge, wsL, wsR, lout, rout);
    }
    mps_label_kernel<<<dim3(B_TOT), dim3(32), 0, stream>>>(lab, lout, rout, (float*)d_out);
}
